// DualAttention_26096221290898
// MI455X (gfx1250) — compile-verified, matrix operand order checked by reading, not run on hardware
//
#include <hip/hip_runtime.h>

#define B_SZ   1024
#define L_SZ   199
#define P_SZ   200      // L+1 rows (keys/queries)
#define D_SZ   128
#define DIM_SZ 256
#define NITER  50
#define MT     4        // M-subtiles per wave in K2 (64 rows per block)

typedef __attribute__((ext_vector_type(16))) __bf16 v16bf;
typedef __attribute__((ext_vector_type(8)))  __bf16 v8bf;
typedef __attribute__((ext_vector_type(8)))  float  v8f;

__device__ __forceinline__ unsigned short f2bf(float f) {
    unsigned u = __float_as_uint(f);
    u += 0x7FFFu + ((u >> 16) & 1u);           // round-to-nearest-even
    return (unsigned short)(u >> 16);
}
__device__ __forceinline__ float bf2f(unsigned short h) {
    return __uint_as_float(((unsigned)h) << 16);
}
// Fragment = 8 bf16 at p[0..7] and 8 bf16 at p[16..23]; write halves directly.
__device__ __forceinline__ v16bf ld_frag(const unsigned short* p) {
    v16bf a;
    *(v8bf*)&a                     = *(const v8bf*)p;
    *(v8bf*)((char*)&a + 16)       = *(const v8bf*)(p + 16);
    return a;
}

// ---------------------------------------------------------------- K0: weights
// Wk/Wv [256][128] f32 -> transposed bf16 [128][256] so B-fragments are
// contiguous 16B chunks along K.
__global__ __launch_bounds__(256) void wprep_kernel(
    const float* __restrict__ Wk, const float* __restrict__ Wv,
    unsigned short* __restrict__ wkT, unsigned short* __restrict__ wvT) {
    int blk = blockIdx.x;
    const float* src = (blk < D_SZ) ? Wk : Wv;
    unsigned short* dst = (blk < D_SZ) ? wkT : wvT;
    int n = (blk < D_SZ) ? blk : blk - D_SZ;
    int c = threadIdx.x;
    dst[n * DIM_SZ + c] = f2bf(src[c * D_SZ + n]);
}

// ------------------------------------------------------- K1: mean emb + alpha
__global__ __launch_bounds__(128) void mean_alpha_kernel(
    const int* __restrict__ x, const int* __restrict__ pos,
    const float* __restrict__ item_emb, const float* __restrict__ pos_emb,
    const float* __restrict__ wa, const float* __restrict__ ba,
    float* __restrict__ mean_ws, float* __restrict__ alpha_ws) {
    __shared__ float red[128];
    int b = blockIdx.x, d = threadIdx.x;
    float acc = 0.f;
    for (int l = 0; l < L_SZ; ++l) {
        int id = x[b * L_SZ + l];
        if (id != 0) acc += item_emb[(size_t)id * D_SZ + d];
    }
    float mean = acc * (1.0f / (float)L_SZ);
    mean_ws[b * D_SZ + d] = mean;
    int plast = pos[b * P_SZ + (P_SZ - 1)];
    float pe = pos_emb[plast * D_SZ + d];
    float part = mean * wa[d] + pe * wa[D_SZ + d];
    red[d] = part; __syncthreads();
#pragma unroll
    for (int s = 64; s > 0; s >>= 1) { if (d < s) red[d] += red[d + s]; __syncthreads(); }
    if (d == 0) {
        float t = red[0] + ba[0];
        float a = 1.0f / (1.0f + __expf(-t)) + 1.0f;
        if (a == 1.0f) a = 1.00001f;
        alpha_ws[b] = a;
    }
}

// ------------------------------------------- K2: K/V projection, bf16 WMMA
// One block = (batch, 64-row tile). 8 waves; wave w owns N-tile [16w,16w+16).
// Each wave accumulates MT=4 M-subtiles so every B fragment feeds 4 WMMAs
// (x2 for K and V sharing the A fragment): 64 WMMAs per wave.
__global__ __launch_bounds__(256) void kv_wmma_kernel(
    const int* __restrict__ x, const int* __restrict__ pos,
    const float* __restrict__ item_emb, const float* __restrict__ pos_emb,
    const float* __restrict__ mean_ws,
    const unsigned short* __restrict__ wkT, const unsigned short* __restrict__ wvT,
    const float* __restrict__ bk, const float* __restrict__ bv,
    unsigned short* __restrict__ k_ws, unsigned short* __restrict__ v_ws) {
    __shared__ __attribute__((aligned(16))) unsigned short sA[16 * MT][DIM_SZ + 8];
    int b = blockIdx.y;
    int tile = blockIdx.x;             // 64-row tile index, 0..3
    int tid = threadIdx.x;
    int c = tid;                       // column 0..255 of A tile

    // prefetch L2-resident weight images (emits global_prefetch_b8)
    __builtin_prefetch(wkT + (size_t)tid * 128, 0, 3);
    __builtin_prefetch(wvT + (size_t)tid * 128, 0, 3);

    // build 64x256 bf16 activation tile by gather
#pragma unroll 1
    for (int r = 0; r < 16 * MT; ++r) {
        int R = tile * (16 * MT) + r;
        float v = 0.f;
        if (R < P_SZ) {
            if (c < D_SZ) {
                v = (R < L_SZ) ? item_emb[(size_t)x[b * L_SZ + R] * D_SZ + c]
                               : mean_ws[b * D_SZ + c];
            } else {
                v = pos_emb[pos[b * P_SZ + R] * D_SZ + (c - D_SZ)];
            }
        }
        sA[r][c] = f2bf(v);
    }
    __syncthreads();

    int wave = tid >> 5;
    int lane = tid & 31;
    int nt = wave * 16;
    int mrow = lane & 15;              // A row within subtile held by this lane
    int khalf = (lane >> 4) * 8;       // lanes 16-31 hold K offset +8 / +24
    int n = nt + mrow;                 // output column for C fragment

    v8f ck[MT] = {}; v8f cv[MT] = {};
#pragma unroll
    for (int ks = 0; ks < 8; ++ks) {
        int k0 = ks * 32 + khalf;
        v16bf bk_ = ld_frag(&wkT[n * DIM_SZ + k0]);
        v16bf bv_ = ld_frag(&wvT[n * DIM_SZ + k0]);
#pragma unroll
        for (int mt = 0; mt < MT; ++mt) {
            v16bf a = ld_frag(&sA[mt * 16 + mrow][k0]);
            ck[mt] = __builtin_amdgcn_wmma_f32_16x16x32_bf16(false, a, false, bk_,
                                                             (short)0, ck[mt], false, false);
            cv[mt] = __builtin_amdgcn_wmma_f32_16x16x32_bf16(false, a, false, bv_,
                                                             (short)0, cv[mt], false, false);
        }
    }
    // epilogue: C layout  lane<16 -> (M=j,N=lane), lane>=16 -> (M=8+j,N=lane-16)
    float bkn = bk[n], bvn = bv[n];
    int mbase = (lane >> 4) * 8;
#pragma unroll
    for (int mt = 0; mt < MT; ++mt) {
#pragma unroll
        for (int j = 0; j < 8; ++j) {
            int R = tile * (16 * MT) + mt * 16 + mbase + j;
            if (R < P_SZ) {
                size_t row = (size_t)(b * P_SZ + R);
                k_ws[row * D_SZ + n] = f2bf(fmaxf(ck[mt][j] + bkn, 0.f));
                v_ws[row * D_SZ + n] = f2bf(fmaxf(cv[mt][j] + bvn, 0.f));
            }
        }
    }
}

// ------------------------------------------------- K3: last-row Q projection
__global__ __launch_bounds__(128) void qlast_kernel(
    const int* __restrict__ pos, const float* __restrict__ pos_emb,
    const float* __restrict__ mean_ws, const float* __restrict__ Wq,
    const float* __restrict__ bq, float* __restrict__ q_ws) {
    __shared__ float xl[DIM_SZ];
    int b = blockIdx.x, d = threadIdx.x;
    xl[d] = mean_ws[b * D_SZ + d];
    xl[D_SZ + d] = pos_emb[pos[b * P_SZ + (P_SZ - 1)] * D_SZ + d];
    __syncthreads();
    float acc = bq[d];
#pragma unroll 4
    for (int cc = 0; cc < DIM_SZ; ++cc) acc += xl[cc] * Wq[cc * D_SZ + d];
    q_ws[b * D_SZ + d] = fmaxf(acc, 0.f);
}

// ---------------------------------- block reductions (all 256 lanes active)
__device__ __forceinline__ float block_reduce_max(float v, float* red, int tid) {
    red[tid] = v; __syncthreads();
#pragma unroll
    for (int s = 128; s > 0; s >>= 1) { if (tid < s) red[tid] = fmaxf(red[tid], red[tid + s]); __syncthreads(); }
    float r = red[0]; __syncthreads();
    return r;
}
__device__ __forceinline__ float block_reduce_sum(float v, float* red, int tid) {
    red[tid] = v; __syncthreads();
#pragma unroll
    for (int s = 128; s > 0; s >>= 1) { if (tid < s) red[tid] += red[tid + s]; __syncthreads(); }
    float r = red[0]; __syncthreads();
    return r;
}

// --------------------- K4: scores -> entmax bisect -> weighted V -> normalize
__global__ __launch_bounds__(256) void attn_kernel(
    const int* __restrict__ x,
    const unsigned short* __restrict__ k_ws, const unsigned short* __restrict__ v_ws,
    const float* __restrict__ q_ws, const float* __restrict__ alpha_ws,
    float* __restrict__ out) {
    __shared__ float sq[D_SZ];
    __shared__ float sW[256];
    __shared__ float red[256];
    int b = blockIdx.x, tid = threadIdx.x;
    if (tid < D_SZ) sq[tid] = q_ws[b * D_SZ + tid];
    __syncthreads();

    float alpha = alpha_ws[b];
    float am1 = alpha - 1.0f;
    float e = 1.0f / am1;

    float Xa = -INFINITY;                         // padding lanes contribute p=0
    if (tid < P_SZ) {
        const uint4* kr4 = (const uint4*)&k_ws[(size_t)(b * P_SZ + tid) * D_SZ];
        float acc = 0.f;
#pragma unroll
        for (int cc = 0; cc < D_SZ / 8; ++cc) {   // b128 loads, unpack 8 bf16
            uint4 w = kr4[cc];
            int c8 = cc * 8;
            acc += sq[c8+0]*__uint_as_float(w.x << 16) + sq[c8+1]*__uint_as_float(w.x & 0xffff0000u);
            acc += sq[c8+2]*__uint_as_float(w.y << 16) + sq[c8+3]*__uint_as_float(w.y & 0xffff0000u);
            acc += sq[c8+4]*__uint_as_float(w.z << 16) + sq[c8+5]*__uint_as_float(w.z & 0xffff0000u);
            acc += sq[c8+6]*__uint_as_float(w.w << 16) + sq[c8+7]*__uint_as_float(w.w & 0xffff0000u);
        }
        float s = acc * 0.08838834764831843f;     // 1/sqrt(128)
        bool valid = (tid == P_SZ - 1) || (x[b * L_SZ + tid] != 0);
        Xa = valid ? s * am1 : -INFINITY;
    }

    float maxv = block_reduce_max(Xa, red, tid);
    float tau_lo = maxv - 1.0f;
    float tau_hi = maxv - __expf(am1 * __logf(1.0f / (float)P_SZ));
    float zlo = Xa - tau_lo;
    float plo = (zlo > 0.f) ? __expf(e * __logf(zlo)) : 0.f;
    float f_lo = block_reduce_sum(plo, red, tid) - 1.0f;

    float dm = tau_hi - tau_lo;
    float t_lo = tau_lo, t_m = tau_lo;
    for (int it = 0; it < NITER; ++it) {
        dm *= 0.5f;
        t_m = t_lo + dm;
        float z = Xa - t_m;
        float pm = (z > 0.f) ? __expf(e * __logf(z)) : 0.f;
        float f_m = block_reduce_sum(pm, red, tid) - 1.0f;   // block-uniform
        if (f_m * f_lo >= 0.f) t_lo = t_m;                   // non-divergent
    }
    float z = Xa - t_m;
    float p = (z > 0.f) ? __expf(e * __logf(z)) : 0.f;
    float S = block_reduce_sum(p, red, tid);
    sW[tid] = p / S;
    __syncthreads();

    // attw @ V : 256 threads, each owns channel d and half of the key range
    int d = tid & (D_SZ - 1);
    int kh = tid >> 7;                 // 0 or 1
    const unsigned short* vb = &v_ws[(size_t)(b * P_SZ) * D_SZ + d];
    float attp = 0.f;
#pragma unroll 4
    for (int k = kh * (P_SZ / 2); k < (kh + 1) * (P_SZ / 2); ++k)
        attp += sW[k] * bf2f(vb[(size_t)k * D_SZ]);
    red[tid] = attp; __syncthreads();
    float att = 0.f;
    if (tid < D_SZ) att = fmaxf(red[tid] + red[tid + D_SZ], 0.f);
    __syncthreads();
    float n2 = block_reduce_sum((tid < D_SZ) ? att * att : 0.f, red, tid);
    if (tid < D_SZ) {
        float nrm = fmaxf(sqrtf(n2), 1e-12f);
        out[b * D_SZ + tid] = att / nrm;
    }
}

extern "C" void kernel_launch(void* const* d_in, const int* in_sizes, int n_in,
                              void* d_out, int out_size, void* d_ws, size_t ws_size,
                              hipStream_t stream) {
    const int*   x        = (const int*)d_in[0];
    const int*   pos      = (const int*)d_in[1];
    const float* item_emb = (const float*)d_in[2];
    const float* pos_emb  = (const float*)d_in[3];
    const float* Wq = (const float*)d_in[4];
    const float* bq = (const float*)d_in[5];
    const float* Wk = (const float*)d_in[6];
    const float* bk = (const float*)d_in[7];
    const float* Wv = (const float*)d_in[8];
    const float* bv = (const float*)d_in[9];
    const float* wa = (const float*)d_in[10];
    const float* ba = (const float*)d_in[11];
    float* out = (float*)d_out;

    char* ws = (char*)d_ws;
    size_t off = 0;
    float* mean_ws  = (float*)(ws + off);              off += (size_t)B_SZ * D_SZ * 4;
    float* alpha_ws = (float*)(ws + off);              off += (size_t)B_SZ * 4;
    float* q_ws     = (float*)(ws + off);              off += (size_t)B_SZ * D_SZ * 4;
    unsigned short* wkT  = (unsigned short*)(ws + off); off += (size_t)D_SZ * DIM_SZ * 2;
    unsigned short* wvT  = (unsigned short*)(ws + off); off += (size_t)D_SZ * DIM_SZ * 2;
    unsigned short* k_ws = (unsigned short*)(ws + off); off += (size_t)B_SZ * P_SZ * D_SZ * 2;
    unsigned short* v_ws = (unsigned short*)(ws + off); off += (size_t)B_SZ * P_SZ * D_SZ * 2;

    wprep_kernel<<<2 * D_SZ, DIM_SZ, 0, stream>>>(Wk, Wv, wkT, wvT);
    mean_alpha_kernel<<<B_SZ, D_SZ, 0, stream>>>(x, pos, item_emb, pos_emb, wa, ba,
                                                 mean_ws, alpha_ws);
    dim3 g2((P_SZ + 16 * MT - 1) / (16 * MT), B_SZ);   // 4 x 1024
    kv_wmma_kernel<<<g2, 256, 0, stream>>>(x, pos, item_emb, pos_emb, mean_ws,
                                           wkT, wvT, bk, bv, k_ws, v_ws);
    qlast_kernel<<<B_SZ, D_SZ, 0, stream>>>(pos, pos_emb, mean_ws, Wq, bq, q_ws);
    attn_kernel<<<B_SZ, 256, 0, stream>>>(x, k_ws, v_ws, q_ws, alpha_ws, out);
}
